// LSTM_36103495090751
// MI455X (gfx1250) — matrix operand order repaired and checked by reading, not compiled, not run on hardware
//
#include <hip/hip_runtime.h>
#include <hip/hip_bf16.h>
#include <hip/hip_fp16.h>

typedef _Float16 half_t;
typedef __attribute__((ext_vector_type(16))) _Float16 v16h;
typedef __attribute__((ext_vector_type(8)))  _Float16 v8h;
typedef __attribute__((ext_vector_type(8)))  float    v8f;
typedef __attribute__((ext_vector_type(4)))  unsigned v4u;
typedef __attribute__((ext_vector_type(4)))  int      v4i;
typedef __attribute__((ext_vector_type(8)))  int      v8i;

#define B_BATCH  256
#define T_SEQ    512
#define D_IN0    128
#define H_DIM    512
#define FOURH    2048
#define NBLK     16      // one block per 32 hidden units; all 16 co-resident -> grid sync safe
#define NTHREADS 512     // 16 wave32 waves; wave w owns batch rows [16w, 16w+16)

#ifndef __has_builtin
#define __has_builtin(x) 0
#endif

#if __has_builtin(__builtin_amdgcn_tensor_load_to_lds)
#define HAVE_TDM 1
#else
#define HAVE_TDM 0
#endif

// Hardware transcendental tanh (v_tanh_f32) if the toolchain exposes it.
#if __has_builtin(__builtin_amdgcn_tanhf)
static __device__ __forceinline__ float fast_tanh(float x) {
  return __builtin_amdgcn_tanhf(x);
}
#else
static __device__ __forceinline__ float fast_tanh(float x) { return tanhf(x); }
#endif

static __device__ __forceinline__ float fast_sigmoid(float x) {
  // sigmoid(x) = 0.5*tanh(x/2) + 0.5 : one TRANS op (co-executes with WMMA)
  return fmaf(0.5f, fast_tanh(0.5f * x), 0.5f);
}

static __device__ __forceinline__ v16h pack16(v8h lo, v8h hi) {
  v16h r;
#pragma unroll
  for (int i = 0; i < 8; ++i) { r[i] = lo[i]; r[i + 8] = hi[i]; }
  return r;
}

static __device__ __forceinline__ v8f wmma_f16(v16h a, v16h b, v8f c) {
  // D(f32) = A(16x32 f16) * B(32x16 f16) + C(f32)
  return __builtin_amdgcn_wmma_f32_16x16x32_f16(false, a, false, b, (short)0, c,
                                                false, false);
}

#if HAVE_TDM
// TDM 2D tile load: `rows` x 512 halfs from global (row stride 512 halfs) into
// LDS with hardware padding of 4 DWORDs (8 halfs) after every 256 DWORDs
// (one 512-half row) -> LDS row stride 520 halfs, matching the WMMA B layout.
static __device__ __forceinline__ void tdm_load_whh(unsigned lds_addr,
                                                    const half_t* gptr,
                                                    unsigned rows) {
  const unsigned long long ga = (unsigned long long)(uintptr_t)gptr;
  v4u g0;
  g0[0] = 1u;                                              // count=1 (valid D#)
  g0[1] = lds_addr;                                        // lds_addr [63:32]
  g0[2] = (unsigned)(ga & 0xffffffffu);                    // global_addr[31:0]
  g0[3] = (unsigned)((ga >> 32) & 0x1ffffffu) | (2u << 30);// addr[56:32] | type=2
  v8i g1;
  g1[0] = (int)((1u << 16)      // data_size = 2 bytes
                | (1u << 20)    // pad_enable
                | (7u << 22)    // pad_interval: 256 DWORDs (= one 1024B row)
                | (3u << 25));  // pad_amount: 4 DWORDs (= 8 halfs)
  g1[1] = (int)((unsigned)H_DIM << 16);   // tensor_dim0 = 512 (low 16)
  g1[2] = (int)(rows << 16);              // dim0 hi=0 | tensor_dim1 = rows
  g1[3] = (int)((unsigned)H_DIM << 16);   // dim1 hi=0 | tile_dim0 = 512
  g1[4] = (int)rows;                      // tile_dim1 = rows | tile_dim2 = 0
  g1[5] = H_DIM;                          // tensor_dim0_stride = 512 (low 32)
  g1[6] = 0;                              // stride hi | tensor_dim1_stride lo
  g1[7] = 0;
  const v4i gz = {0, 0, 0, 0};
#if __clang_major__ >= 23
  const v8i gz8 = {0, 0, 0, 0, 0, 0, 0, 0};
  __builtin_amdgcn_tensor_load_to_lds(g0, g1, gz, gz, gz8, 0);
#else
  __builtin_amdgcn_tensor_load_to_lds(g0, g1, gz, gz, 0);
#endif
}
#endif  // HAVE_TDM

__global__ void k_cast_f16(const float* __restrict__ s, half_t* __restrict__ d, int n) {
  int i = blockIdx.x * blockDim.x + threadIdx.x;
  int stride = gridDim.x * blockDim.x;
  for (; i < n; i += stride) d[i] = (half_t)s[i];
}

__global__ void k_bias_sum(const float* __restrict__ a, const float* __restrict__ b,
                           float* __restrict__ o, int n) {
  int i = blockIdx.x * blockDim.x + threadIdx.x;
  if (i < n) o[i] = a[i] + b[i];
}

// One fused LSTM layer. Block `blk` owns hidden units [32*blk, 32*blk+32):
// its 128 gate columns (i,f,g,o) of W_hh stay resident in LDS for all T steps.
// Each wave computes a 16-row batch tile across those 128 gate columns
// (8 WMMA N-tiles, fp32 accumulators seeded with the fused bias).
template <int DIN, bool IS_L1>
__global__ __launch_bounds__(NTHREADS, 1) void lstm_layer_k(
    const half_t* __restrict__ xin,   // [B][T][DIN] f16 (layer2: = out_seq of layer1)
    const half_t* __restrict__ Wih,   // [4H][DIN] f16
    const half_t* __restrict__ Whh,   // [4H][H]   f16
    const float* __restrict__ bias,   // [4H] fused b_ih + b_hh
    half_t* __restrict__ out_seq,     // layer1: [B][T][H] output (doubles as h history)
    half_t* __restrict__ hA,          // layer2: h ping buffer [B][H]
    half_t* __restrict__ hB,          // layer2: h pong buffer [B][H]
    unsigned* __restrict__ sync_ctr,  // grid step-barrier counter (zeroed by host memset)
    const float* __restrict__ Wfc,    // [H]  (layer2 epilogue)
    const float* __restrict__ bfc,    // [1]
    float* __restrict__ yout)         // [B]
{
  constexpr int WSTR = H_DIM + 8;  // padded LDS row stride (halves) -> bank spread
  __shared__ half_t whh_lds[128 * WSTR];  // 133,120 B

  const int blk  = blockIdx.x;
  const int j0   = blk * 32;
  const int tid  = threadIdx.x;
  const int wave = tid >> 5;
  const int lane = tid & 31;
  const int lh   = lane & 15;  // column / row-within-tile index
  const int sel  = lane >> 4;  // which K/M half this lane covers

  // ---- Stage this block's W_hh slice into LDS (once; reused for 512 steps) ----
#if HAVE_TDM
  if (wave == 0) {
    const unsigned lds_base = (unsigned)(uintptr_t)&whh_lds[0];
#pragma unroll
    for (int gate = 0; gate < 4; ++gate) {
      tdm_load_whh(lds_base + (unsigned)(gate * 32) * (WSTR * 2),
                   Whh + (size_t)(gate * H_DIM + j0) * H_DIM, 32);
    }
    __builtin_amdgcn_s_wait_tensorcnt(0);
  }
#else
  for (int idx = tid; idx < 128 * (H_DIM / 8); idx += NTHREADS) {
    const int lr   = idx / (H_DIM / 8);
    const int kk   = (idx % (H_DIM / 8)) * 8;
    const int gate = lr >> 5;
    const int unit = lr & 31;
    const v8h v = *(const v8h*)(Whh + (size_t)(gate * H_DIM + j0 + unit) * H_DIM + kk);
    *(v8h*)(&whh_lds[lr * WSTR + kk]) = v;
  }
#endif
  __syncthreads();

  // Bias is a function of the gate column only -> one scalar per N-tile per lane.
  float bias_r[8];
#pragma unroll
  for (int tt = 0; tt < 8; ++tt) {
    const int gate = tt >> 1, g = tt & 1;
    bias_r[tt] = bias[gate * H_DIM + j0 + g * 16 + lh];
  }

  // Cell state: this lane's fixed (row,unit) elements of the D fragment.
  float c_reg[2][8];
#pragma unroll
  for (int g = 0; g < 2; ++g)
#pragma unroll
    for (int r = 0; r < 8; ++r) c_reg[g][r] = 0.0f;

  const int brow = wave * 16 + lh;  // batch row whose A-fragment this lane loads

  for (int t = 0; t < T_SEQ; ++t) {
    v8f acc[8];
#pragma unroll
    for (int tt = 0; tt < 8; ++tt) {
#pragma unroll
      for (int r = 0; r < 8; ++r) acc[tt][r] = bias_r[tt];
    }

    // ---- input contribution: x_t @ W_ih^T (W_ih streamed, L2-resident) ----
    {
      const half_t* arow = xin + ((size_t)brow * T_SEQ + t) * DIN;
#pragma unroll 1
      for (int kc = 0; kc < DIN; kc += 32) {
        const v16h a = pack16(*(const v8h*)(arow + kc + sel * 8),
                              *(const v8h*)(arow + kc + 16 + sel * 8));
#pragma unroll
        for (int tt = 0; tt < 8; ++tt) {
          const int gate = tt >> 1, g = tt & 1;
          const half_t* bp =
              Wih + (size_t)(gate * H_DIM + j0 + g * 16 + lh) * DIN + kc + sel * 8;
          const v16h b = pack16(*(const v8h*)bp, *(const v8h*)(bp + 16));
          acc[tt] = wmma_f16(a, b, acc[tt]);
        }
      }
    }

    // ---- recurrent contribution: h_{t-1} @ W_hh^T (B from LDS). h_0 = 0. ----
    if (t > 0) {
      const half_t* hrow;
      if (IS_L1)
        hrow = out_seq + ((size_t)brow * T_SEQ + (t - 1)) * H_DIM;
      else
        hrow = ((t & 1) ? hA : hB) + (size_t)brow * H_DIM;
#pragma unroll 1
      for (int kc = 0; kc < H_DIM; kc += 32) {
        const v16h a = pack16(*(const v8h*)(hrow + kc + sel * 8),
                              *(const v8h*)(hrow + kc + 16 + sel * 8));
#pragma unroll
        for (int tt = 0; tt < 8; ++tt) {
          const int gate = tt >> 1, g = tt & 1;
          const half_t* bp = &whh_lds[(gate * 32 + g * 16 + lh) * WSTR + kc + sel * 8];
          const v16h b = pack16(*(const v8h*)bp, *(const v8h*)(bp + 16));
          acc[tt] = wmma_f16(a, b, acc[tt]);
        }
      }
    }

    // ---- gate nonlinearities + cell/hidden update (c stays in registers) ----
    half_t* hnext = IS_L1 ? (half_t*)nullptr : ((t & 1) ? hB : hA);
#pragma unroll
    for (int g = 0; g < 2; ++g) {
      const int u = g * 16 + lh;  // unit within this block's 32
#pragma unroll
      for (int r = 0; r < 8; ++r) {
        const int row = wave * 16 + sel * 8 + r;  // D-fragment row mapping
        const float iv = fast_sigmoid(acc[0 + g][r]);
        const float fv = fast_sigmoid(acc[2 + g][r]);
        const float gv = fast_tanh(acc[4 + g][r]);
        const float ov = fast_sigmoid(acc[6 + g][r]);
        const float c  = fv * c_reg[g][r] + iv * gv;
        c_reg[g][r]    = c;
        const float h  = ov * fast_tanh(c);
        const half_t h16 = (half_t)h;
        if (IS_L1)
          out_seq[((size_t)row * T_SEQ + t) * H_DIM + j0 + u] = h16;
        else
          hnext[(size_t)row * H_DIM + j0 + u] = h16;
      }
    }

    // ---- device-wide step barrier (16 co-resident blocks) ----
    __threadfence();
    __syncthreads();
    if (tid == 0) {
      __hip_atomic_fetch_add(sync_ctr, 1u, __ATOMIC_RELEASE, __HIP_MEMORY_SCOPE_AGENT);
      const unsigned tgt = (unsigned)(t + 1) * NBLK;
      while (__hip_atomic_load(sync_ctr, __ATOMIC_ACQUIRE, __HIP_MEMORY_SCOPE_AGENT) < tgt)
        __builtin_amdgcn_s_sleep(2);
    }
    __syncthreads();
    __threadfence();
  }

  // ---- epilogue (layer 2 only): y = relu(relu(h_T) @ W_fc^T + b_fc) ----
  if (!IS_L1 && blk == 0) {
    const half_t* hlast = hB;  // t = 511 (odd) wrote hB
    for (int b = tid; b < B_BATCH; b += NTHREADS) {
      float s = bfc[0];
      for (int j = 0; j < H_DIM; ++j) {
        float hv = (float)hlast[(size_t)b * H_DIM + j];
        hv = hv > 0.0f ? hv : 0.0f;
        s += hv * Wfc[j];
      }
      yout[b] = s > 0.0f ? s : 0.0f;
    }
  }
}

extern "C" void kernel_launch(void* const* d_in, const int* in_sizes, int n_in,
                              void* d_out, int out_size, void* d_ws, size_t ws_size,
                              hipStream_t stream) {
  (void)in_sizes; (void)n_in; (void)out_size; (void)ws_size;
  const float* x    = (const float*)d_in[0];
  const float* Wih1 = (const float*)d_in[1];
  const float* Whh1 = (const float*)d_in[2];
  const float* bih1 = (const float*)d_in[3];
  const float* bhh1 = (const float*)d_in[4];
  const float* Wih2 = (const float*)d_in[5];
  const float* Whh2 = (const float*)d_in[6];
  const float* bih2 = (const float*)d_in[7];
  const float* bhh2 = (const float*)d_in[8];
  const float* Wfc  = (const float*)d_in[9];
  const float* bfc  = (const float*)d_in[10];
  float* y = (float*)d_out;

  char* ws = (char*)d_ws;
  size_t off = 0;
  auto alloc = [&](size_t bytes) {
    char* p = ws + off;
    off += (bytes + 255) & ~(size_t)255;
    return p;
  };
  unsigned* ctrs = (unsigned*)alloc(256);  // [0]=layer1, [1]=layer2 step counters
  half_t* xf16 = (half_t*)alloc((size_t)B_BATCH * T_SEQ * D_IN0 * 2);
  half_t* o1   = (half_t*)alloc((size_t)B_BATCH * T_SEQ * H_DIM * 2);
  half_t* wih1 = (half_t*)alloc((size_t)FOURH * D_IN0 * 2);
  half_t* whh1 = (half_t*)alloc((size_t)FOURH * H_DIM * 2);
  half_t* wih2 = (half_t*)alloc((size_t)FOURH * H_DIM * 2);
  half_t* whh2 = (half_t*)alloc((size_t)FOURH * H_DIM * 2);
  float*  b1   = (float*)alloc(FOURH * 4);
  float*  b2   = (float*)alloc(FOURH * 4);
  half_t* hA   = (half_t*)alloc((size_t)B_BATCH * H_DIM * 2);
  half_t* hB   = (half_t*)alloc((size_t)B_BATCH * H_DIM * 2);

  hipMemsetAsync(ctrs, 0, 256, stream);  // capture-legal; deterministic per call

  auto cast = [&](const float* s, half_t* d, int n) {
    int blocks = (n + 255) / 256;
    if (blocks > 32768) blocks = 32768;
    k_cast_f16<<<blocks, 256, 0, stream>>>(s, d, n);
  };
  cast(x,    xf16, B_BATCH * T_SEQ * D_IN0);
  cast(Wih1, wih1, FOURH * D_IN0);
  cast(Whh1, whh1, FOURH * H_DIM);
  cast(Wih2, wih2, FOURH * H_DIM);
  cast(Whh2, whh2, FOURH * H_DIM);
  k_bias_sum<<<(FOURH + 255) / 256, 256, 0, stream>>>(bih1, bhh1, b1, FOURH);
  k_bias_sum<<<(FOURH + 255) / 256, 256, 0, stream>>>(bih2, bhh2, b2, FOURH);

  lstm_layer_k<D_IN0, true><<<NBLK, NTHREADS, 0, stream>>>(
      xf16, wih1, whh1, b1, o1, nullptr, nullptr, &ctrs[0], nullptr, nullptr, nullptr);
  lstm_layer_k<H_DIM, false><<<NBLK, NTHREADS, 0, stream>>>(
      o1, wih2, whh2, b2, nullptr, hA, hB, &ctrs[1], Wfc, bfc, y);
}
